// InsideOutsideMultiLayerEncoder_29678224015896
// MI455X (gfx1250) — matrix operand order repaired and checked by reading, not compiled, not run on hardware
//
#include <hip/hip_runtime.h>
#include <hip/hip_bf16.h>
#include <cstdint>

// ---------------------------------------------------------------------------
// Fully fused tree-encoder for MI455X (gfx1250, wave32, WMMA bf16).
// One workgroup (256 thr = 8 waves) owns 64 tokens (32 pairs) and carries them
// through all 4 transformer layers + head entirely in LDS (~310 KB / WGP).
// Weights are pre-converted to bf16 in d_ws (L2-resident) and streamed into
// LDS by the Tensor Data Mover (tensor_load_to_lds + s_wait_tensorcnt),
// double-buffered so the DMA overlaps the v_wmma_f32_16x16x32_bf16 phase.
// ---------------------------------------------------------------------------

typedef __attribute__((ext_vector_type(16))) __bf16 v16bf;
typedef __attribute__((ext_vector_type(8)))  float  v8f;
typedef __attribute__((ext_vector_type(4)))  unsigned int uint32x4;
typedef __attribute__((ext_vector_type(8)))  int int32x8;
typedef __attribute__((ext_vector_type(4)))  int int32x4;

#if __has_builtin(__builtin_amdgcn_tensor_load_to_lds) && __has_builtin(__builtin_amdgcn_s_wait_tensorcnt)
#define USE_TDM 1
#else
#define USE_TDM 0
#endif

#define TOK 64
#define D   256
#define NCHUNK 128
#define KSTEP  32
#define WPAD   40            // LDS row pitch of weight tile (32 data + 8 pad bf16)
#define TILE_EL (128 * WPAD) // 5120 bf16 = 10240 B per staged tile
#define XBS 264              // padded stride for bf16 activation rows (d=256)
#define H1S 1032             // padded stride for bf16 FFN hidden rows (ff=1024)

// LDS layout (bytes)
#define XF_OFF   0           // float [64][256]          65536
#define YF_OFF   65536       // float [64][256]          65536
#define XB_OFF   131072      // bf16  [64][XBS]          33792
#define BIG_OFF  164864      // bf16  qkv/o | h1 | hL    132096
#define WST_OFF  296960      // bf16  2 x TILE_EL        20480
#define SMEM_BYTES 317440
#define OBASE 49152          // o offset inside big (bf16 elements, after 64x768 qkv)

// bf16 weight offsets inside d_ws (elements)
#define WQKV_OFF 0u          // 4 x 768*256
#define WO_OFF   786432u     // 4 x 256*256
#define W1_OFF   1048576u    // 4 x 1024*256
#define W2_OFF   2097152u    // 4 x 256*1024
#define WL1_OFF  3145728u
#define WL2_OFF  3211264u
#define WR1_OFF  3244032u
#define WR2_OFF  3309568u
#define WTOTAL   3342336u

__device__ __forceinline__ float gelu_exact(float x) {
  return 0.5f * x * (1.0f + erff(x * 0.70710678118654752f));
}
__device__ __forceinline__ uint32_t pk_bf16(float a, float b) {
  union { __bf16 h[2]; uint32_t u; } c;
  c.h[0] = (__bf16)a; c.h[1] = (__bf16)b;
  return c.u;
}
__device__ __forceinline__ float bflo(uint32_t u) { return __uint_as_float(u << 16); }
__device__ __forceinline__ float bfhi(uint32_t u) { return __uint_as_float(u & 0xffff0000u); }

union U4 { uint4 v; uint32_t u[4]; };

#if USE_TDM
// TDM: DMA a 2D tile (tile_dim0=32 k-elems, tile_dim1=128 n-rows) of bf16 from
// a row-major [N][K] global matrix into LDS with 8-elem row padding (pitch 40).
__device__ __forceinline__ void tdm_load_tile(const __bf16* gsrc, const __bf16* ldst,
                                              int N, int K)
{
  const uint64_t ga  = (uint64_t)(uintptr_t)gsrc;
  const uint32_t lds = (uint32_t)(uintptr_t)ldst;   // LDS aperture: low 32 bits
  uint32x4 g0;
  g0[0] = 1u;                                          // count=1, user descriptor
  g0[1] = lds;                                         // lds_addr (bytes)
  g0[2] = (uint32_t)ga;                                // global_addr[31:0]
  g0[3] = (uint32_t)((ga >> 32) & 0x01ffffffu) | 0x80000000u; // addr[56:32] | type=2
  int32x8 g1;
  g1[0] = (int)((1u << 16)      // data_size = 2 bytes
              | (1u << 20)      // pad_enable
              | (3u << 22)      // pad_interval: 16 DWORDs (32 bf16)
              | (3u << 25));    // pad_amount:   4 DWORDs  (8 bf16)
  g1[1] = (int)(((uint32_t)K & 0xffffu) << 16);                          // tensor_dim0 lo
  g1[2] = (int)((((uint32_t)K >> 16) & 0xffffu) | (((uint32_t)N & 0xffffu) << 16));
  g1[3] = (int)((((uint32_t)N >> 16) & 0xffffu) | (32u << 16));          // tile_dim0=32
  g1[4] = 128;                                                           // tile_dim1=128
  g1[5] = K;                                                             // dim0 stride lo
  g1[6] = 0;
  g1[7] = 0;
  const int32x4 z4 = {0, 0, 0, 0};
#if __clang_major__ >= 23
  const int32x8 z8 = {0, 0, 0, 0, 0, 0, 0, 0};
  __builtin_amdgcn_tensor_load_to_lds(g0, g1, z4, z4, z8, 0);
#else
  __builtin_amdgcn_tensor_load_to_lds(g0, g1, z4, z4, 0);
#endif
}
#endif

// C = act(A(64xK bf16 LDS, stride lda) @ W(NxK bf16 global)^T + bias)
// EPI: 0 = bf16 out, 1 = gelu -> bf16 out, 2 = f32 out
template <int EPI, int N, int K>
__device__ void gemm_tile(const __bf16* __restrict__ A, int lda,
                          const __bf16* __restrict__ Wb,
                          const float* __restrict__ bias,
                          __bf16* outb, int ldo,
                          float* outf, int ldof,
                          __bf16* wst, int tid)
{
  const int lane = tid & 31;
  const int w    = tid >> 5;
  const int wm   = w >> 1;      // 0..3 -> rows [wm*16, +16)
  const int wn   = w & 1;       // 0..1 -> cols [wn*64, +64)
  const int l15  = lane & 15;
  const int lh   = lane >> 4;   // 0/1

#if USE_TDM
  const bool issuer = (w == 0);
  int buf = 0;
  if (issuer) {
    tdm_load_tile(Wb, wst, N, K);
    __builtin_amdgcn_s_wait_tensorcnt(0);
  }
  __syncthreads();
#else
  int sn[4], sk4[4];
#pragma unroll
  for (int j = 0; j < 4; ++j) {
    const int lin = tid + j * 256;
    sn[j]  = lin >> 3;           // tile row 0..127
    sk4[j] = (lin & 7) << 2;     // k offset 0,4,..,28
  }
#endif

  for (int nb = 0; nb < N; nb += NCHUNK) {
    v8f acc[4];
#pragma unroll
    for (int t = 0; t < 4; ++t)
#pragma unroll
      for (int r = 0; r < 8; ++r) acc[t][r] = 0.0f;

#if !USE_TDM
    uint2 g[4];
#pragma unroll
    for (int j = 0; j < 4; ++j)
      g[j] = *(const uint2*)(Wb + (nb + sn[j]) * K + sk4[j]);
#endif

    for (int kb = 0; kb < K; kb += KSTEP) {
#if USE_TDM
      // issue DMA for the next tile into the other buffer (overlaps WMMA)
      int nnb = nb, nkb = kb + KSTEP;
      if (nkb == K) { nkb = 0; nnb += NCHUNK; }
      if (issuer && nnb < N)
        tdm_load_tile(Wb + (size_t)nnb * K + nkb, wst + (buf ^ 1) * TILE_EL, N, K);
      const __bf16* cw = wst + buf * TILE_EL;
#else
      __syncthreads();
#pragma unroll
      for (int j = 0; j < 4; ++j)
        *(uint2*)(wst + sn[j] * WPAD + sk4[j]) = g[j];
      // prefetch next tile's registers (overlaps WMMA phase)
      int nnb = nb, nkb = kb + KSTEP;
      if (nkb == K) { nkb = 0; nnb += NCHUNK; }
      if (nnb < N) {
#pragma unroll
        for (int j = 0; j < 4; ++j)
          g[j] = *(const uint2*)(Wb + (nnb + sn[j]) * K + nkb + sk4[j]);
      }
      __syncthreads();
      const __bf16* cw = wst;
#endif

      // A fragment: 16-bit A 16x32 layout (lane m = l&15; K halves by lane>>4)
      const __bf16* pa = A + (wm * 16 + l15) * lda + kb + lh * 8;
      union { uint4 q[2]; v16bf v; } ua;
      ua.q[0] = *(const uint4*)(pa);
      ua.q[1] = *(const uint4*)(pa + 16);

#pragma unroll
      for (int t = 0; t < 4; ++t) {
        const __bf16* pb = cw + (wn * 64 + t * 16 + l15) * WPAD + lh * 8;
        union { uint4 q[2]; v16bf v; } ub;
        ub.q[0] = *(const uint4*)(pb);
        ub.q[1] = *(const uint4*)(pb + 16);
        acc[t] = __builtin_amdgcn_wmma_f32_16x16x32_bf16(
                   false, ua.v, false, ub.v, (short)0, acc[t], false, false);
      }

#if USE_TDM
      if (issuer) __builtin_amdgcn_s_wait_tensorcnt(0);
      __syncthreads();
      buf ^= 1;
#endif
    }

    // epilogue: C layout lane l, reg r -> row wm*16 + (l>>4)*8 + r, col l&15
    const int rbase = wm * 16 + lh * 8;
#pragma unroll
    for (int t = 0; t < 4; ++t) {
      const int col = nb + wn * 64 + t * 16 + l15;
      const float bv = bias[col];
#pragma unroll
      for (int r = 0; r < 8; ++r) {
        float v = acc[t][r] + bv;
        if (EPI == 1) v = gelu_exact(v);
        const int row = rbase + r;
        if (EPI == 2) outf[row * ldof + col] = v;
        else          outb[row * ldo  + col] = (__bf16)v;
      }
    }
  }
#if !USE_TDM
  __syncthreads();   // protect wst before next gemm
#endif
}

// x = InstanceNorm(x + y) over last dim (256); refresh padded bf16 mirror.
__device__ void inorm_update(float* xf, const float* yf, __bf16* xb, int tid) {
  const int lane = tid & 31, w = tid >> 5;
  const int j0 = lane * 8;
  for (int r = w; r < TOK; r += 8) {
    const float4 xa = *(const float4*)(xf + r * D + j0);
    const float4 xc = *(const float4*)(xf + r * D + j0 + 4);
    const float4 ya = *(const float4*)(yf + r * D + j0);
    const float4 yc = *(const float4*)(yf + r * D + j0 + 4);
    float v[8] = { xa.x + ya.x, xa.y + ya.y, xa.z + ya.z, xa.w + ya.w,
                   xc.x + yc.x, xc.y + yc.y, xc.z + yc.z, xc.w + yc.w };
    float s = 0.f, s2 = 0.f;
#pragma unroll
    for (int i = 0; i < 8; ++i) { s += v[i]; s2 += v[i] * v[i]; }
#pragma unroll
    for (int off = 16; off > 0; off >>= 1) {
      s  += __shfl_xor(s,  off, 32);
      s2 += __shfl_xor(s2, off, 32);
    }
    const float mean = s * (1.f / 256.f);
    const float rstd = rsqrtf(s2 * (1.f / 256.f) - mean * mean + 1e-5f);
#pragma unroll
    for (int i = 0; i < 8; ++i) v[i] = (v[i] - mean) * rstd;
    *(float4*)(xf + r * D + j0)     = make_float4(v[0], v[1], v[2], v[3]);
    *(float4*)(xf + r * D + j0 + 4) = make_float4(v[4], v[5], v[6], v[7]);
    *(uint4*)(xb + r * XBS + j0) =
        make_uint4(pk_bf16(v[0], v[1]), pk_bf16(v[2], v[3]),
                   pk_bf16(v[4], v[5]), pk_bf16(v[6], v[7]));
  }
}

// S=2 attention within pairs; qkv rows of 768 bf16 (q|k|v, each H=4 x dh=64).
__device__ void attention_pairs(const __bf16* qkv, __bf16* o, int tid) {
  const int p = tid >> 3;        // 32 pairs
  const int h = (tid >> 1) & 3;  // 4 heads
  const int s = tid & 1;         // query token
  const __bf16* qv = qkv + (2 * p + s) * 768 +       h * 64;
  const __bf16* k0 = qkv + (2 * p + 0) * 768 + 256 + h * 64;
  const __bf16* k1 = qkv + (2 * p + 1) * 768 + 256 + h * 64;
  const __bf16* v0 = qkv + (2 * p + 0) * 768 + 512 + h * 64;
  const __bf16* v1 = qkv + (2 * p + 1) * 768 + 512 + h * 64;
  float s0 = 0.f, s1 = 0.f;
#pragma unroll 2
  for (int j = 0; j < 64; j += 8) {
    U4 q, a, b;
    q.v = *(const uint4*)(qv + j);
    a.v = *(const uint4*)(k0 + j);
    b.v = *(const uint4*)(k1 + j);
#pragma unroll
    for (int t = 0; t < 4; ++t) {
      const float ql = bflo(q.u[t]), qh = bfhi(q.u[t]);
      s0 += ql * bflo(a.u[t]) + qh * bfhi(a.u[t]);
      s1 += ql * bflo(b.u[t]) + qh * bfhi(b.u[t]);
    }
  }
  s0 *= 0.125f; s1 *= 0.125f;              // 1/sqrt(64)
  const float m  = fmaxf(s0, s1);
  const float e0 = __expf(s0 - m), e1 = __expf(s1 - m);
  const float inv = 1.f / (e0 + e1);
  const float a0 = e0 * inv, a1 = e1 * inv;
  __bf16* op = o + (2 * p + s) * XBS + h * 64;
#pragma unroll 2
  for (int j = 0; j < 64; j += 8) {
    U4 a, b, r;
    a.v = *(const uint4*)(v0 + j);
    b.v = *(const uint4*)(v1 + j);
#pragma unroll
    for (int t = 0; t < 4; ++t) {
      const float lo = a0 * bflo(a.u[t]) + a1 * bflo(b.u[t]);
      const float hi = a0 * bfhi(a.u[t]) + a1 * bfhi(b.u[t]);
      r.u[t] = pk_bf16(lo, hi);
    }
    *(uint4*)(op + j) = r.v;
  }
}

// weight pre-conversion: fp32 -> bf16 (flat copy)
__global__ __launch_bounds__(256)
void cvt_weights(const float* __restrict__ src, __bf16* __restrict__ dst, int n4) {
  const int i = blockIdx.x * 256 + threadIdx.x;
  if (i < n4) {
    const float4 v = *(const float4*)(src + i * 4);
    *(uint2*)(dst + i * 4) = make_uint2(pk_bf16(v.x, v.y), pk_bf16(v.z, v.w));
  }
}

__global__ __launch_bounds__(256, 1)
void tree_encoder_fused(
    const float* __restrict__ children, const float* __restrict__ roles,
    const __bf16* __restrict__ Wb,
    const float* __restrict__ bqkv, const float* __restrict__ bo,
    const float* __restrict__ b1,   const float* __restrict__ b2,
    const float* __restrict__ bl1,  const float* __restrict__ bl2,
    const float* __restrict__ br1,  const float* __restrict__ br2,
    const float* __restrict__ ln_g, const float* __restrict__ ln_b,
    float* __restrict__ out, int npairs_total)
{
  extern __shared__ char smem[];
  float*  xf  = (float*) (smem + XF_OFF);
  float*  yf  = (float*) (smem + YF_OFF);
  __bf16* xb  = (__bf16*)(smem + XB_OFF);
  __bf16* big = (__bf16*)(smem + BIG_OFF);
  __bf16* wst = (__bf16*)(smem + WST_OFF);

  const int tid  = threadIdx.x;
  const int tok0 = (int)blockIdx.x * TOK;

  // x = children + role embedding (row 0/1 by token parity)
  for (int it = tid; it < TOK * D / 4; it += 256) {
    const int r = it >> 6, j4 = (it & 63) << 2;
    const float4 c  = *(const float4*)(children + ((tok0 + r) * D + j4));
    const float4 ro = *(const float4*)(roles + ((r & 1) * D + j4));
    const float4 v  = make_float4(c.x + ro.x, c.y + ro.y, c.z + ro.z, c.w + ro.w);
    *(float4*)(xf + r * D + j4) = v;
    *(uint2*)(xb + r * XBS + j4) = make_uint2(pk_bf16(v.x, v.y), pk_bf16(v.z, v.w));
  }
  __syncthreads();

  for (int l = 0; l < 4; ++l) {
    // qkv = x @ Wqkv^T + b  -> big[64][768] bf16
    gemm_tile<0, 768, 256>(xb, XBS, Wb + WQKV_OFF + l * 196608u, bqkv + l * 768,
                           big, 768, nullptr, 0, wst, tid);
    __syncthreads();
    attention_pairs(big, big + OBASE, tid);
    __syncthreads();
    // yf = o @ Wo^T + bo
    gemm_tile<2, 256, 256>(big + OBASE, XBS, Wb + WO_OFF + l * 65536u, bo + l * 256,
                           nullptr, 0, yf, D, wst, tid);
    __syncthreads();
    inorm_update(xf, yf, xb, tid);
    __syncthreads();
    // h1 = gelu(x @ W1^T + b1) -> big[64][1032-padded] bf16
    gemm_tile<1, 1024, 256>(xb, XBS, Wb + W1_OFF + l * 262144u, b1 + l * 1024,
                            big, H1S, nullptr, 0, wst, tid);
    __syncthreads();
    // yf = h1 @ W2^T + b2
    gemm_tile<2, 256, 1024>(big, H1S, Wb + W2_OFF + l * 262144u, b2 + l * 256,
                            nullptr, 0, yf, D, wst, tid);
    __syncthreads();
    inorm_update(xf, yf, xb, tid);
    __syncthreads();
  }

  // Head: left constituent MLP (even rows used), lc -> yf[:,0:128]
  gemm_tile<1, 256, 256>(xb, XBS, Wb + WL1_OFF, bl1, big, XBS, nullptr, 0, wst, tid);
  __syncthreads();
  gemm_tile<2, 128, 256>(big, XBS, Wb + WL2_OFF, bl2, nullptr, 0, yf, 256, wst, tid);
  __syncthreads();
  // right constituent MLP (odd rows used), rc -> yf[:,128:256]
  gemm_tile<1, 256, 256>(xb, XBS, Wb + WR1_OFF, br1, big, XBS, nullptr, 0, wst, tid);
  __syncthreads();
  gemm_tile<2, 128, 256>(big, XBS, Wb + WR2_OFF, br2, nullptr, 0, yf + 128, 256, wst, tid);
  __syncthreads();

  // mat_scores[p] = dot(lc[2p], rc[2p+1]) / sqrt(128)
  {
    const int p = tid >> 3, gi = tid & 7;
    const float4* l4 = (const float4*)(yf + (2 * p) * D + gi * 16);
    const float4* r4 = (const float4*)(yf + (2 * p + 1) * D + 128 + gi * 16);
    float sum = 0.f;
#pragma unroll
    for (int t = 0; t < 4; ++t) {
      const float4 a = l4[t], b = r4[t];
      sum += a.x * b.x + a.y * b.y + a.z * b.z + a.w * b.w;
    }
    sum += __shfl_xor(sum, 1, 32);
    sum += __shfl_xor(sum, 2, 32);
    sum += __shfl_xor(sum, 4, 32);
    if (gi == 0) out[(int)blockIdx.x * 32 + p] = sum * 0.08838834764831845f;
  }
  __syncthreads();

  // src = LayerNorm(x[2p] + x[2p+1]) * g + b
  {
    const int lane = tid & 31, w = tid >> 5;
    float* src = out + npairs_total;
    const int j0 = lane * 8;
    const float4 g0 = *(const float4*)(ln_g + j0);
    const float4 g1 = *(const float4*)(ln_g + j0 + 4);
    const float4 be0 = *(const float4*)(ln_b + j0);
    const float4 be1 = *(const float4*)(ln_b + j0 + 4);
    for (int p = w; p < 32; p += 8) {
      const float4 a0 = *(const float4*)(xf + (2 * p) * D + j0);
      const float4 a1 = *(const float4*)(xf + (2 * p) * D + j0 + 4);
      const float4 b0 = *(const float4*)(xf + (2 * p + 1) * D + j0);
      const float4 b1v = *(const float4*)(xf + (2 * p + 1) * D + j0 + 4);
      float v[8] = { a0.x + b0.x, a0.y + b0.y, a0.z + b0.z, a0.w + b0.w,
                     a1.x + b1v.x, a1.y + b1v.y, a1.z + b1v.z, a1.w + b1v.w };
      float s = 0.f, s2 = 0.f;
#pragma unroll
      for (int i = 0; i < 8; ++i) { s += v[i]; s2 += v[i] * v[i]; }
#pragma unroll
      for (int off = 16; off > 0; off >>= 1) {
        s  += __shfl_xor(s,  off, 32);
        s2 += __shfl_xor(s2, off, 32);
      }
      const float mean = s * (1.f / 256.f);
      const float rstd = rsqrtf(s2 * (1.f / 256.f) - mean * mean + 1e-5f);
      const int P = (int)blockIdx.x * 32 + p;
      *(float4*)(src + P * D + j0) =
          make_float4((v[0] - mean) * rstd * g0.x + be0.x,
                      (v[1] - mean) * rstd * g0.y + be0.y,
                      (v[2] - mean) * rstd * g0.z + be0.z,
                      (v[3] - mean) * rstd * g0.w + be0.w);
      *(float4*)(src + P * D + j0 + 4) =
          make_float4((v[4] - mean) * rstd * g1.x + be1.x,
                      (v[5] - mean) * rstd * g1.y + be1.y,
                      (v[6] - mean) * rstd * g1.z + be1.z,
                      (v[7] - mean) * rstd * g1.w + be1.w);
    }
  }
}

extern "C" void kernel_launch(void* const* d_in, const int* in_sizes, int n_in,
                              void* d_out, int out_size, void* d_ws, size_t ws_size,
                              hipStream_t stream)
{
  const float* children = (const float*)d_in[0];
  const float* roles    = (const float*)d_in[1];
  const float* bqkv     = (const float*)d_in[3];
  const float* bo       = (const float*)d_in[5];
  const float* b1       = (const float*)d_in[7];
  const float* b2       = (const float*)d_in[9];
  const float* bl1      = (const float*)d_in[11];
  const float* bl2      = (const float*)d_in[13];
  const float* br1      = (const float*)d_in[15];
  const float* br2      = (const float*)d_in[17];
  const float* ln_g     = (const float*)d_in[18];
  const float* ln_b     = (const float*)d_in[19];

  __bf16* Wb = (__bf16*)d_ws;

  // pre-convert all weights fp32 -> bf16 into d_ws (L2-resident, 6.4 MB)
  struct { int src_idx; unsigned dst_off; unsigned count; } segs[8] = {
    { 2,  WQKV_OFF, 786432u },  // Wqkv (4 layers)
    { 4,  WO_OFF,   262144u },  // Wo
    { 6,  W1_OFF,   1048576u }, // W1
    { 8,  W2_OFF,   1048576u }, // W2
    { 10, WL1_OFF,  65536u },
    { 12, WL2_OFF,  32768u },
    { 14, WR1_OFF,  65536u },
    { 16, WR2_OFF,  32768u },
  };
  for (int i = 0; i < 8; ++i) {
    const int n4 = (int)(segs[i].count / 4);
    cvt_weights<<<(n4 + 255) / 256, 256, 0, stream>>>(
        (const float*)d_in[segs[i].src_idx], Wb + segs[i].dst_off, n4);
  }

  const int tokens = in_sizes[0] / D;   // 131072
  const int blocks = tokens / TOK;      // 2048
  const int npairs = tokens / 2;        // 65536

  (void)n_in; (void)out_size; (void)ws_size;

  hipFuncSetAttribute((const void*)tree_encoder_fused,
                      hipFuncAttributeMaxDynamicSharedMemorySize, SMEM_BYTES);

  tree_encoder_fused<<<blocks, 256, SMEM_BYTES, stream>>>(
      children, roles, Wb, bqkv, bo, b1, b2, bl1, bl2, br1, br2,
      ln_g, ln_b, (float*)d_out, npairs);
}
